// TFPerformerAttention_31671088840960
// MI455X (gfx1250) — matrix operand order repaired and checked by reading, not compiled, not run on hardware
//
#include <hip/hip_runtime.h>
#include <math.h>

// ---------------- problem constants ----------------
#define B_   2
#define L_   2048
#define H_   16
#define DM_  1024
#define D_   64
#define M_   266     // random features
#define MP   288     // padded to 9*32 for bf16 WMMA K-steps
#define ROWS (B_*H_*L_)   // 65536
#define BL   (B_*L_)      // 4096

__device__ __constant__ float kS4   = 0.17677669529663687f;  // DM^-0.25
__device__ __constant__ float kCM   = 0.06131393394849658f;  // M^-0.5
__device__ __constant__ float kEPS  = 1e-4f;

typedef __attribute__((ext_vector_type(16))) __bf16 bf16x16;
typedef __attribute__((ext_vector_type(8)))  float  f32x8;
typedef __attribute__((ext_vector_type(4)))  unsigned int u32x4;
typedef __attribute__((ext_vector_type(8)))  int i32x8;
typedef __attribute__((ext_vector_type(4)))  int i32x4;

union FragB {
  bf16x16 v;
  unsigned int u[8];
  uint4 q[2];
};

__device__ __forceinline__ unsigned int pack2bf(float a, float b) {
  union { __bf16 h; unsigned short s; } x, y;
  x.h = (__bf16)a; y.h = (__bf16)b;
  return (unsigned int)x.s | ((unsigned int)y.s << 16);
}
// bf16 -> f32 is a 16-bit shift (no convert op needed)
__device__ __forceinline__ float bflo(unsigned int p) { return __uint_as_float(p << 16); }
__device__ __forceinline__ float bfhi(unsigned int p) { return __uint_as_float(p & 0xffff0000u); }

// ---------------- WMMA (wave32, 16x16x32 bf16 -> f32) ----------------
__device__ __forceinline__ f32x8 wmma_bf16(bf16x16 a, bf16x16 b, f32x8 c) {
  return __builtin_amdgcn_wmma_f32_16x16x32_bf16(false, a, false, b, (short)0, c,
                                                 false, false);
}

// A fragment, row-major source: two contiguous 8-elem runs -> two b128 loads.
__device__ __forceinline__ bf16x16 frag_a_v(const __bf16* base, int stride, int lane) {
  int m = lane & 15, half = lane >> 4;
  FragB f;
  f.q[0] = *(const uint4*)(base + m * stride + half * 16);
  f.q[1] = *(const uint4*)(base + m * stride + half * 16 + 8);
  return f.v;
}
// A fragment from f32 global with scale; vectorized float4 loads.
__device__ __forceinline__ bf16x16 frag_a_f32s(const float* base, int stride,
                                               float sc, int lane) {
  int m = lane & 15, half = lane >> 4;
  const float* r0 = base + m * stride + half * 16;
  const float* r1 = base + m * stride + half * 16 + 8;
  float4 f0 = *(const float4*)r0, f1 = *(const float4*)(r0 + 4);
  float4 g0 = *(const float4*)r1, g1 = *(const float4*)(r1 + 4);
  bf16x16 a;
  a[0]=(__bf16)(f0.x*sc); a[1]=(__bf16)(f0.y*sc); a[2]=(__bf16)(f0.z*sc); a[3]=(__bf16)(f0.w*sc);
  a[4]=(__bf16)(f1.x*sc); a[5]=(__bf16)(f1.y*sc); a[6]=(__bf16)(f1.z*sc); a[7]=(__bf16)(f1.w*sc);
  a[8]=(__bf16)(g0.x*sc); a[9]=(__bf16)(g0.y*sc); a[10]=(__bf16)(g0.z*sc); a[11]=(__bf16)(g0.w*sc);
  a[12]=(__bf16)(g1.x*sc); a[13]=(__bf16)(g1.y*sc); a[14]=(__bf16)(g1.z*sc); a[15]=(__bf16)(g1.w*sc);
  return a;
}
// B fragment from K-pair-interleaved source: one aligned b32 per dword.
__device__ __forceinline__ bf16x16 frag_b_pair(const unsigned int* basew, int Wd,
                                               int kp0, int n0, int lane) {
  int n = lane & 15, kbp = (lane >> 4) * 8;
  FragB f;
#pragma unroll
  for (int u = 0; u < 8; ++u)
    f.u[u] = basew[(kp0 + kbp + u) * Wd + n0 + n];
  return f.v;
}
// B fragment from row-major source transposed (pairs contiguous in source row).
__device__ __forceinline__ bf16x16 frag_bt_pairs(const unsigned int* rows, int Wp,
                                                 int row0, int kp0, int lane) {
  int n = lane & 15, kbp = (lane >> 4) * 8;
  FragB f;
#pragma unroll
  for (int u = 0; u < 8; ++u)
    f.u[u] = rows[(row0 + n) * Wp + kp0 + kbp + u];
  return f.v;
}
// B fragment from pair-interleaved f32 source (state S): float2 per pair.
__device__ __forceinline__ bf16x16 frag_b_f32p(const float2* basep, int Wd,
                                               int kp0, int n0, int lane) {
  int n = lane & 15, kbp = (lane >> 4) * 8;
  bf16x16 b;
#pragma unroll
  for (int u = 0; u < 8; ++u) {
    float2 f = basep[(kp0 + kbp + u) * Wd + n0 + n];
    b[2*u]   = (__bf16)f.x;
    b[2*u+1] = (__bf16)f.y;
  }
  return b;
}

// ---- Tensor Data Mover: contiguous 2D bf16 tile (rows x cols) -> LDS --------
// D# per CDNA5 ISA 8.3/8.4; 2D so groups 2/3/extra are zero. 6-arg builtin
// (clang-23 / therock-10.0 signature).
__device__ __forceinline__ void tdm_load_2d_bf16(unsigned int lds_off,
                                                 const void* gaddr,
                                                 unsigned int rows,
                                                 unsigned int cols) {
  unsigned long long ga = (unsigned long long)(size_t)gaddr;
  u32x4 g0;
  g0[0] = 1u;                                              // count=1 (valid), user
  g0[1] = lds_off;                                         // lds_addr [63:32]
  g0[2] = (unsigned int)ga;                                // global_addr [95:64]
  g0[3] = ((unsigned int)(ga >> 32) & 0x01ffffffu) | (2u << 30);  // addr hi, type=2
  i32x8 g1;
  g1[0] = (int)(1u << 16);                 // wg_mask=0, data_size=1 (2 bytes)
  g1[1] = (int)((cols & 0xffffu) << 16);   // tensor_dim0 lo16 @ bits 63:48
  g1[2] = (int)((cols >> 16) | ((rows & 0xffffu) << 16));  // dim0 hi / dim1 lo
  g1[3] = (int)((rows >> 16) | ((cols & 0xffffu) << 16));  // dim1 hi / tile_dim0
  g1[4] = (int)(rows & 0xffffu);           // tile_dim1 = rows, tile_dim2 = 0
  g1[5] = (int)cols;                       // tensor_dim0_stride lo32
  g1[6] = 0;
  g1[7] = 0;
  __builtin_amdgcn_tensor_load_to_lds(g0, g1, (i32x4)0, (i32x4)0, (i32x8)0, 0);
}

// ---------------- 1) dense GEMM + bias: C[Mr,N] = A[Mr,K] @ W[K,N] + b -------
__global__ __launch_bounds__(256) void gemm_bias_f32(
    const float* __restrict__ A, const float* __restrict__ W,
    const float* __restrict__ bias, float* __restrict__ C,
    int Mr, int N, int K) {
  __shared__ __bf16 As[128][32];
  __shared__ unsigned int Bsw[16][128];   // 32x128 bf16, K-pair interleaved
  int tid = threadIdx.x, lane = tid & 31, w = tid >> 5;
  int row0 = blockIdx.y * 128, col0 = blockIdx.x * 128;
  int wr = (w & 3) * 32;        // wave rows (2 tiles)
  int wc = (w >> 2) * 64;       // wave cols (4 tiles)
  f32x8 acc[2][4] = {};
  for (int kk = 0; kk < K; kk += 32) {
#pragma unroll
    for (int i = 0; i < 4; ++i) {               // stage A: float4 -> packed b64
      int idx = tid + i * 256;
      int r = idx >> 3, g = idx & 7;            // row, 4-col group
      float4 f = *(const float4*)&A[(size_t)(row0 + r) * K + kk + g * 4];
      uint2 o; o.x = pack2bf(f.x, f.y); o.y = pack2bf(f.z, f.w);
      *(uint2*)&As[r][g * 4] = o;
    }
#pragma unroll
    for (int i = 0; i < 2; ++i) {               // stage B: 2x float4 -> b128
      int idx = tid + i * 256;
      int pr = idx >> 5, g = idx & 31;
      const float* w0 = &W[(size_t)(kk + 2 * pr) * N + col0 + g * 4];
      float4 f0 = *(const float4*)w0;
      float4 f1 = *(const float4*)(w0 + N);
      uint4 o; o.x = pack2bf(f0.x, f1.x); o.y = pack2bf(f0.y, f1.y);
              o.z = pack2bf(f0.z, f1.z); o.w = pack2bf(f0.w, f1.w);
      *(uint4*)&Bsw[pr][g * 4] = o;
    }
    if (kk + 32 < K) {   // prefetch next K tile -> global_prefetch_b8
      __builtin_prefetch(&A[(size_t)(row0 + (tid >> 1)) * K + kk + 32 + (tid & 1) * 16], 0, 3);
      __builtin_prefetch(&W[(size_t)(kk + 32 + (tid >> 3)) * N + col0 + (tid & 7) * 16], 0, 3);
    }
    __syncthreads();
    bf16x16 a0 = frag_a_v(&As[wr][0], 32, lane);
    bf16x16 a1 = frag_a_v(&As[wr + 16][0], 32, lane);
#pragma unroll
    for (int j = 0; j < 4; ++j) {
      bf16x16 bf = frag_b_pair(&Bsw[0][0], 128, 0, wc + j * 16, lane);
      acc[0][j] = wmma_bf16(a0, bf, acc[0][j]);
      acc[1][j] = wmma_bf16(a1, bf, acc[1][j]);
    }
    __syncthreads();
  }
  int n = lane & 15, half = lane >> 4;
#pragma unroll
  for (int i = 0; i < 2; ++i)
#pragma unroll
    for (int j = 0; j < 4; ++j)
#pragma unroll
      for (int v = 0; v < 8; ++v) {
        int r = row0 + wr + i * 16 + v + 8 * half;
        int c = col0 + wc + j * 16 + n;
        C[(size_t)r * N + c] = acc[i][j][v] + bias[c];
      }
}

// ---------------- 2) h_k and global stabilizer (min of ||k~||^2) -------------
__global__ void prep_hk(const float* __restrict__ Kf, float* __restrict__ hk,
                        unsigned int* __restrict__ minbits) {
  int r = blockIdx.x * blockDim.x + threadIdx.x;
  if (r >= ROWS) return;
  int bh = r / L_, l = r % L_;
  int b = bh / H_, h = bh % H_;
  const float4* kp4 = (const float4*)(Kf + ((size_t)(b * L_ + l)) * DM_ + h * D_);
  float s = 0.f;
#pragma unroll
  for (int d = 0; d < 16; ++d) {
    float4 f = kp4[d];
    s += f.x * f.x + f.y * f.y + f.z * f.z + f.w * f.w;
  }
  s *= kS4 * kS4;
  hk[r] = -0.5f * s;
  atomicMin(minbits, __float_as_uint(s));   // s >= 0: uint order == float order
}

// ---------------- 3) random features -> pair-interleaved [32 pairs][MP] ------
__global__ void conv_rf(const float* __restrict__ rf, unsigned int* __restrict__ rfTw) {
  int i = blockIdx.x * blockDim.x + threadIdx.x;
  if (i >= 32 * MP) return;
  int pr = i / MP, m = i % MP;
  float lo = 0.f, hi = 0.f;
  if (m < M_) { lo = rf[m * D_ + 2 * pr]; hi = rf[m * D_ + 2 * pr + 1]; }
  rfTw[i] = pack2bf(lo, hi);
}

// ---------------- 4) FAVOR projection + exp kernel ---------------------------
__global__ __launch_bounds__(256) void favor_proj(
    const float* __restrict__ Xf, const unsigned int* __restrict__ rfTw,
    const float* __restrict__ hk, const unsigned int* __restrict__ minbits,
    __bf16* __restrict__ out, int is_query) {
  int tid = threadIdx.x, lane = tid & 31, w = tid >> 5;
  int row0 = blockIdx.y * 64 + (w & 3) * 16;        // 16 rows, same (b,h)
  int col0 = blockIdx.x * 32 + (w >> 2) * 16;
  int bh = row0 / L_, l0 = row0 % L_;
  int b = bh / H_, h = bh % H_;
  const float* abase = Xf + ((size_t)(b * L_ + l0)) * DM_ + h * D_;
  f32x8 acc = {};
#pragma unroll
  for (int ks = 0; ks < D_; ks += 32) {
    bf16x16 a  = frag_a_f32s(abase + ks, DM_, kS4, lane);
    bf16x16 bb = frag_b_pair(rfTw, MP, ks >> 1, col0, lane);
    acc = wmma_bf16(a, bb, acc);
  }
  float kstab = -0.5f * __uint_as_float(*minbits);
  int n = lane & 15, half = lane >> 4;
#pragma unroll
  for (int v = 0; v < 8; ++v) {
    int r = row0 + v + 8 * half;
    int m = col0 + n;
    float o = 0.f;
    if (m < M_) {
      float bias = is_query ? 0.f : (hk[r] - kstab);
      o = kCM * (__expf(acc[v] + bias) + kEPS);
    }
    out[(size_t)r * MP + m] = (__bf16)o;
  }
}

// ---------------- 5) chunked causal linear attention -------------------------
__global__ __launch_bounds__(256) void causal_scan(
    const __bf16* __restrict__ qp, const __bf16* __restrict__ kp,
    const float* __restrict__ Vf, float* __restrict__ ctx) {
  extern __shared__ char smem[];
  float*  S    = (float*)smem;                    // MP*64, (m,d)->(m>>1)*128+2d+(m&1)
  float*  Af   = S + MP * 64;                     // 64*64
  float*  z    = Af + 64 * 64;                    // MP
  float*  dsum = z + MP;                          // 64
  __bf16* Qc   = (__bf16*)(dsum + 64);            // 64*MP row-major
  __bf16* Kc   = Qc + 64 * MP;                    // 64*MP row-major
  __bf16* KcT  = Kc + 64 * MP;                    // MP*64 (transpose of Kc)
  unsigned int* Vcw = (unsigned int*)(KcT + MP * 64);  // 32 pairs x 64
  __bf16* Ab   = (__bf16*)(Vcw + 32 * 64);        // 64*64 row-major

  int tid = threadIdx.x, lane = tid & 31, w = tid >> 5;
  int bh = blockIdx.x, b = bh / H_, h = bh % H_;
  size_t R0 = (size_t)bh * L_;
  unsigned int ldsQ = (unsigned int)(size_t)(void*)Qc;   // LDS byte offsets
  unsigned int ldsK = (unsigned int)(size_t)(void*)Kc;
  for (int i = tid; i < MP * 64; i += 256) S[i] = 0.f;
  for (int i = tid; i < MP; i += 256) z[i] = 0.f;
  __syncthreads();

  for (int c0 = 0; c0 < L_; c0 += 64) {
    // ---- TDM: DMA the contiguous 64xMP bf16 Q'/K' chunk into LDS ----
    if (w == 0) {
      tdm_load_2d_bf16(ldsQ, qp + (R0 + c0) * MP, 64u, (unsigned)MP);
      tdm_load_2d_bf16(ldsK, kp + (R0 + c0) * MP, 64u, (unsigned)MP);
    }
    // ---- V chunk, pair-packed (manual: strided + f32->bf16 convert) ----
    for (int i = tid; i < 32 * 16; i += 256) {
      int pr = i >> 4, g = i & 15;
      const float* vb = Vf + ((size_t)(b * L_) + c0 + 2 * pr) * DM_ + h * D_ + g * 4;
      float4 f0 = *(const float4*)vb;
      float4 f1 = *(const float4*)(vb + DM_);
      uint4 o; o.x = pack2bf(f0.x, f1.x); o.y = pack2bf(f0.y, f1.y);
              o.z = pack2bf(f0.z, f1.z); o.w = pack2bf(f0.w, f1.w);
      *(uint4*)&Vcw[pr * 64 + g * 4] = o;
    }
    if (w == 0) __builtin_amdgcn_s_wait_tensorcnt(0);
    __syncthreads();
    // ---- KcT = transpose(Kc) (dword reads, b16 writes) ----
    {
      const unsigned int* Kw = (const unsigned int*)Kc;
      for (int i = tid; i < 64 * (MP / 2); i += 256) {
        int j = i / (MP / 2), pr = i % (MP / 2);
        union { unsigned int u; __bf16 h[2]; } cc;
        cc.u = Kw[i];
        KcT[(2 * pr) * 64 + j]     = cc.h[0];
        KcT[(2 * pr + 1) * 64 + j] = cc.h[1];
      }
    }
    // ---- A = Qc Kc^T, causal mask (inclusive diag) ----
#pragma unroll
    for (int q = 0; q < 2; ++q) {
      int t = w + q * 8;
      int i0 = (t & 3) * 16, j0 = (t >> 2) * 16;
      f32x8 acc = {};
      for (int ks = 0; ks < MP; ks += 32) {
        bf16x16 a  = frag_a_v(Qc + i0 * MP + ks, MP, lane);
        bf16x16 bb = frag_bt_pairs((const unsigned int*)Kc, MP / 2, j0, ks >> 1, lane);
        acc = wmma_bf16(a, bb, acc);
      }
      int n = lane & 15, half = lane >> 4;
#pragma unroll
      for (int v = 0; v < 8; ++v) {
        int i = i0 + v + 8 * half, j = j0 + n;
        float val = (j <= i) ? acc[v] : 0.f;
        Af[i * 64 + j] = val;
        Ab[i * 64 + j] = (__bf16)val;
      }
    }
    __syncthreads();
    // ---- denominator with z_prev (vectorized reads) ----
    if (tid < 64) {
      float s = 0.f;
      const float4* af4 = (const float4*)(Af + tid * 64);
      for (int j = 0; j < 16; ++j) {
        float4 f = af4[j];
        s += f.x + f.y + f.z + f.w;
      }
      float qz = 0.f;
      const unsigned int* qw = (const unsigned int*)(Qc + tid * MP);
      for (int pr = 0; pr < MP / 2; ++pr) {
        unsigned int pk = qw[pr];
        qz += bflo(pk) * z[2 * pr] + bfhi(pk) * z[2 * pr + 1];
      }
      dsum[tid] = s + qz;
    }
    __syncthreads();
    // ---- z += colsum(Kc), two columns at a time ----
    for (int pr = tid; pr < MP / 2; pr += 256) {
      float a0 = z[2 * pr], a1 = z[2 * pr + 1];
      const unsigned int* Kw = (const unsigned int*)Kc;
      for (int j = 0; j < 64; ++j) {
        unsigned int pk = Kw[j * (MP / 2) + pr];
        a0 += bflo(pk); a1 += bfhi(pk);
      }
      z[2 * pr] = a0; z[2 * pr + 1] = a1;
    }
    // ---- numerator: masked(A) @ Vc + Qc @ S_prev; write ctx ----
#pragma unroll
    for (int q = 0; q < 2; ++q) {
      int t = w + q * 8;
      int i0 = (t & 3) * 16, d0 = (t >> 2) * 16;
      f32x8 acc = {};
      for (int ks = 0; ks < MP; ks += 32) {
        bf16x16 a  = frag_a_v(Qc + i0 * MP + ks, MP, lane);
        bf16x16 bb = frag_b_f32p((const float2*)S, 64, ks >> 1, d0, lane);
        acc = wmma_bf16(a, bb, acc);
      }
#pragma unroll
      for (int ks = 0; ks < 64; ks += 32) {
        bf16x16 a  = frag_a_v(Ab + i0 * 64 + ks, 64, lane);
        bf16x16 bb = frag_b_pair(Vcw, 64, ks >> 1, d0, lane);
        acc = wmma_bf16(a, bb, acc);
      }
      int n = lane & 15, half = lane >> 4;
#pragma unroll
      for (int v = 0; v < 8; ++v) {
        int i = i0 + v + 8 * half;
        ctx[((size_t)(b * L_) + c0 + i) * DM_ + h * D_ + d0 + n] =
            acc[v] / dsum[i];
      }
    }
    __syncthreads();      // all S reads done before S update
    // ---- S += Kc^T @ Vc : 72 tiles, 9/wave; A-frags from KcT (b128 loads) ----
#pragma unroll
    for (int q = 0; q < 9; ++q) {
      int t = w * 9 + q;
      int f0 = (t % 18) * 16, d0 = (t / 18) * 16;
      int n = lane & 15, half = lane >> 4;
      f32x8 cacc;
#pragma unroll
      for (int v = 0; v < 8; ++v) {
        int fm = f0 + v + 8 * half;
        cacc[v] = S[(fm >> 1) * 128 + (d0 + n) * 2 + (fm & 1)];
      }
#pragma unroll
      for (int ks = 0; ks < 64; ks += 32) {
        bf16x16 a  = frag_a_v(KcT + f0 * 64 + ks, 64, lane);
        bf16x16 bb = frag_b_pair(Vcw, 64, ks >> 1, d0, lane);
        cacc = wmma_bf16(a, bb, cacc);
      }
#pragma unroll
      for (int v = 0; v < 8; ++v) {
        int fm = f0 + v + 8 * half;
        S[(fm >> 1) * 128 + (d0 + n) * 2 + (fm & 1)] = cacc[v];
      }
    }
    __syncthreads();      // Kc/Vc/S settled before next chunk overwrite
  }
}

// ---------------- host orchestration ----------------------------------------
extern "C" void kernel_launch(void* const* d_in, const int* in_sizes, int n_in,
                              void* d_out, int out_size, void* d_ws, size_t ws_size,
                              hipStream_t stream) {
  const float* query = (const float*)d_in[0];
  const float* key_  = (const float*)d_in[1];
  const float* value = (const float*)d_in[2];
  const float* Wq = (const float*)d_in[3];  const float* bq = (const float*)d_in[4];
  const float* Wk = (const float*)d_in[5];  const float* bk = (const float*)d_in[6];
  const float* Wv = (const float*)d_in[7];  const float* bv = (const float*)d_in[8];
  const float* Wo = (const float*)d_in[9];  const float* bo = (const float*)d_in[10];
  const float* rf = (const float*)d_in[11];
  float* out = (float*)d_out;

  char* p = (char*)d_ws;
  auto carve = [&](size_t bytes) { char* q = p; p += (bytes + 255) & ~(size_t)255; return q; };
  float*  Qf      = (float*)carve((size_t)BL * DM_ * 4);
  float*  Kf      = (float*)carve((size_t)BL * DM_ * 4);
  float*  Vf      = (float*)carve((size_t)BL * DM_ * 4);
  float*  ctx     = (float*)carve((size_t)BL * DM_ * 4);
  float*  hk      = (float*)carve((size_t)ROWS * 4);
  unsigned int* rfTw = (unsigned int*)carve((size_t)32 * MP * 4);
  __bf16* qprime  = (__bf16*)carve((size_t)ROWS * MP * 2);
  __bf16* kprime  = (__bf16*)carve((size_t)ROWS * MP * 2);
  unsigned int* minbits = (unsigned int*)carve(256);

  dim3 gGemm(DM_ / 128, BL / 128);   // (8, 32)
  gemm_bias_f32<<<gGemm, 256, 0, stream>>>(query, Wq, bq, Qf, BL, DM_, DM_);
  gemm_bias_f32<<<gGemm, 256, 0, stream>>>(key_,  Wk, bk, Kf, BL, DM_, DM_);
  gemm_bias_f32<<<gGemm, 256, 0, stream>>>(value, Wv, bv, Vf, BL, DM_, DM_);

  (void)hipMemsetAsync(minbits, 0x7F, 4, stream);
  prep_hk<<<ROWS / 256, 256, 0, stream>>>(Kf, hk, minbits);
  conv_rf<<<(32 * MP + 255) / 256, 256, 0, stream>>>(rf, rfTw);

  dim3 gProj(MP / 32, ROWS / 64);  // (9, 1024)
  favor_proj<<<gProj, 256, 0, stream>>>(Qf, rfTw, hk, minbits, qprime, 1);
  favor_proj<<<gProj, 256, 0, stream>>>(Kf, rfTw, hk, minbits, kprime, 0);

  size_t smem = (size_t)(MP * 64 + 64 * 64 + MP + 64) * 4   // f32: S, Af, z, dsum
              + (size_t)(2 * 64 * MP + MP * 64 + 64 * 64) * 2  // Qc, Kc, KcT, Ab
              + (size_t)(32 * 64) * 4;                      // Vcw pairs
  causal_scan<<<B_ * H_, 256, smem, stream>>>(qprime, kprime, Vf, ctx);

  gemm_bias_f32<<<gGemm, 256, 0, stream>>>(ctx, Wo, bo, out, BL, DM_, DM_);
}
